// QueryLayer_44805098832334
// MI455X (gfx1250) — matrix operand order repaired and checked by reading, not compiled, not run on hardware
//
#include <hip/hip_runtime.h>
#include <cstdint>
#include <cstddef>

#define B_ 8
#define S_ 1024
#define D_ 128
#define P_ (B_ * S_)
#define EPS_ 1e-7f
#define LN_EPS_ 1e-3f

typedef __bf16 bf16_t;
typedef __attribute__((ext_vector_type(8)))  __bf16 v8bf;
typedef __attribute__((ext_vector_type(16))) __bf16 v16bf;
typedef __attribute__((ext_vector_type(8)))  float  v8f;

#if defined(__has_builtin)
#if __has_builtin(__builtin_amdgcn_tensor_load_to_lds) && \
    __has_builtin(__builtin_amdgcn_s_wait_tensorcnt)
#define HAVE_TDM 1
#endif
#endif
#ifndef HAVE_TDM
#define HAVE_TDM 0
#endif

__device__ __forceinline__ v8f zero8() {
  v8f z = {0.f, 0.f, 0.f, 0.f, 0.f, 0.f, 0.f, 0.f};
  return z;
}

__device__ __forceinline__ v16bf pack16(v8bf lo, v8bf hi) {
  v16bf a;
#pragma unroll
  for (int e = 0; e < 8; ++e) { a[e] = lo[e]; a[e + 8] = hi[e]; }
  return a;
}

// A fragment (16x32 bf16, M x K): lane row = lane&15.
// element e holds K = k0 + 16*half + e
__device__ __forceinline__ v16bf load_a_frag(const bf16_t* rowp, int k0, int half) {
  v8bf lo = *(const v8bf*)(rowp + k0 + 16 * half);
  v8bf hi = *(const v8bf*)(rowp + k0 + 16 * half + 8);
  return pack16(lo, hi);
}

// B fragment (32x16 bf16, K x N): lane col = lane&15, element e holds K = base + e;
// caller passes p = col_base + k0 + 16*half (16 contiguous bf16 in K).
__device__ __forceinline__ v16bf load_b_frag(const bf16_t* p) {
  v8bf lo = *(const v8bf*)p;
  v8bf hi = *(const v8bf*)(p + 8);
  return pack16(lo, hi);
}

// packed bf16 scale: lowers to v_pk_mul_bf16
__device__ __forceinline__ v16bf mul16(v16bf a, bf16_t s) {
  v16bf sv;
#pragma unroll
  for (int e = 0; e < 16; ++e) sv[e] = s;
  return a * sv;
}

__device__ __forceinline__ v8f wmma_bf16(v16bf a, v16bf b, v8f c) {
  return __builtin_amdgcn_wmma_f32_16x16x32_bf16(false, a, false, b, (short)0, c,
                                                 false, false);
}

#if HAVE_TDM
typedef __attribute__((ext_vector_type(4))) unsigned int u32x4;
typedef __attribute__((ext_vector_type(4))) int i32x4;
typedef __attribute__((ext_vector_type(8))) int i32x8;

// TDM 2-D tile load: 128 rows (d) x 32 cols (k_lin), 2-byte elems,
// row stride 16384 elems; lands packed [128][32] at lds_off.
__device__ __forceinline__ void tdm_load_tile(const bf16_t* gsrc, unsigned lds_off) {
  const unsigned long long ga = (unsigned long long)(uintptr_t)gsrc;
  u32x4 g0;
  g0[0] = 1u;                                                  // count=1, user D#
  g0[1] = lds_off;                                             // lds_addr (bytes)
  g0[2] = (unsigned)ga;                                        // global_addr[31:0]
  g0[3] = (unsigned)((ga >> 32) & 0x01ffffffu) | (2u << 30);   // addr[56:32]|type=2
  i32x8 g1 = {
      0x00010000,            // workgroup_mask=0, data_size=1 (2 B)
      (int)(16384u << 16),   // tensor_dim0 = 16384 (k_lin length)
      (int)(128u << 16),     // tensor_dim0 hi | tensor_dim1 = 128 (d rows)
      (int)(32u << 16),      // tensor_dim1 hi | tile_dim0 = 32
      128,                   // tile_dim1 = 128, tile_dim2 = 0
      16384,                 // tensor_dim0_stride = 16384 elems
      0, 0};                 // stride hi | tensor_dim1_stride = 0
  i32x4 gz = {0, 0, 0, 0};
#if __clang_major__ >= 23
  i32x8 gz8 = {0, 0, 0, 0, 0, 0, 0, 0};
  __builtin_amdgcn_tensor_load_to_lds(g0, g1, gz, gz, gz8, 0);
#else
  __builtin_amdgcn_tensor_load_to_lds(g0, g1, gz, gz, 0);
#endif
}
#endif  // HAVE_TDM

// ---------------- prep kernels ----------------

__global__ void k_cast_x(const float* __restrict__ x, bf16_t* __restrict__ xbf,
                         bf16_t* __restrict__ x2bf, bf16_t* __restrict__ xT) {
  const size_t idx = (size_t)blockIdx.x * blockDim.x + threadIdx.x;
  if (idx >= (size_t)P_ * D_) return;
  const float v = x[idx];
  xbf[idx]  = (bf16_t)v;
  x2bf[idx] = (bf16_t)(v * v);
  const size_t pos = idx >> 7;
  const int d  = (int)(idx & (D_ - 1));
  const int bb = (int)(pos >> 10);
  const int s  = (int)(pos & (S_ - 1));
  xT[((size_t)bb * D_ + d) * S_ + s] = (bf16_t)v;
}

__global__ void k_cast_w(const float* __restrict__ Wq, const float* __restrict__ Wm,
                         bf16_t* __restrict__ WqT, bf16_t* __restrict__ WmT) {
  const int idx = blockIdx.x * blockDim.x + threadIdx.x;
  if (idx >= D_ * D_) return;
  const int d = idx >> 7, n = idx & (D_ - 1);
  WqT[n * D_ + d] = (bf16_t)Wq[idx];
  WmT[n * D_ + d] = (bf16_t)Wm[idx];
}

__global__ void k_cast_map(const float* __restrict__ m, bf16_t* __restrict__ mb) {
  const size_t idx = (size_t)blockIdx.x * blockDim.x + threadIdx.x;
  if (idx >= (size_t)D_ * D_ * D_) return;
  mb[idx] = (bf16_t)m[idx];
}

__global__ void k_wc(const float* __restrict__ x, const float* __restrict__ Ww,
                     const float* __restrict__ bw, const float* __restrict__ Wc,
                     const float* __restrict__ bc, float* __restrict__ w,
                     float* __restrict__ c) {
  const int pos = blockIdx.x * blockDim.x + threadIdx.x;
  if (pos >= P_) return;
  const float* xr = x + (size_t)pos * D_;
  float dw = 0.f, dc = 0.f;
  for (int d = 0; d < D_; ++d) { const float v = xr[d]; dw += v * Ww[d]; dc += v * Wc[d]; }
  w[pos] = fmaxf(dw + bw[0], 0.f);
  c[pos] = fmaxf(dc + bc[0], 0.f);
}

// ---------------- q/m projection GEMM (WMMA) ----------------
__global__ void k_qm(const float* __restrict__ bq, const float* __restrict__ bm,
                     const bf16_t* __restrict__ xbf, const bf16_t* __restrict__ WqT,
                     const bf16_t* __restrict__ WmT, bf16_t* __restrict__ a1,
                     bf16_t* __restrict__ mbf, bf16_t* __restrict__ m2bf,
                     float* __restrict__ sumqp, float* __restrict__ nb2) {
  const int lane = threadIdx.x & 31;
  const int wv   = threadIdx.x >> 5;
  const int half = lane >> 4;
  const int lcol = lane & 15;
  const long pos0 = (long)blockIdx.x * 16;

  const bf16_t* arow = xbf + (size_t)(pos0 + lcol) * D_;
  v16bf A[4];
#pragma unroll
  for (int kc = 0; kc < 4; ++kc) A[kc] = load_a_frag(arow, kc * 32, half);

  v8f Cq[2], Cm[2];
#pragma unroll
  for (int t = 0; t < 2; ++t) { Cq[t] = zero8(); Cm[t] = zero8(); }

#pragma unroll
  for (int t = 0; t < 2; ++t) {
    const int n0 = (wv * 2 + t) * 16;
    const bf16_t* qcol = WqT + (size_t)(n0 + lcol) * D_;
    const bf16_t* mcol = WmT + (size_t)(n0 + lcol) * D_;
#pragma unroll
    for (int kc = 0; kc < 4; ++kc) {
      const int k0 = kc * 32 + 16 * half;
      Cq[t] = wmma_bf16(A[kc], load_b_frag(qcol + k0), Cq[t]);
      Cm[t] = wmma_bf16(A[kc], load_b_frag(mcol + k0), Cm[t]);
    }
  }

  float acc1[8], acc2[8];
#pragma unroll
  for (int r = 0; r < 8; ++r) { acc1[r] = 0.f; acc2[r] = 0.f; }

#pragma unroll
  for (int t = 0; t < 2; ++t) {
    const int col = (wv * 2 + t) * 16 + lcol;
    const float bqv = bq[col], bmv = bm[col];
#pragma unroll
    for (int r = 0; r < 8; ++r) {
      const long row = pos0 + r + 8 * half;
      const float qp = Cq[t][r] + bqv + EPS_;
      const float mm = 1.f / (1.f + __expf(-(Cm[t][r] + bmv)));
      const size_t gi = (size_t)row * D_ + col;
      a1[gi]  = (bf16_t)(mm * qp);
      mbf[gi] = (bf16_t)mm;
      m2bf[gi] = (bf16_t)(mm * mm);
      acc1[r] += qp;
      acc2[r] += qp * qp;
    }
  }
#pragma unroll
  for (int r = 0; r < 8; ++r) {
#pragma unroll
    for (int msk = 1; msk <= 8; msk <<= 1) {
      acc1[r] += __shfl_xor(acc1[r], msk, 32);
      acc2[r] += __shfl_xor(acc2[r], msk, 32);
    }
  }
  if (lcol == 0) {
#pragma unroll
    for (int r = 0; r < 8; ++r) {
      atomicAdd(&sumqp[pos0 + r + 8 * half], acc1[r]);
      atomicAdd(&nb2[pos0 + r + 8 * half], acc2[r]);
    }
  }
}

// ---------------- fused causal attention-like pass ----------------
__global__ void k_attn(const bf16_t* __restrict__ a1, const bf16_t* __restrict__ mbf,
                       const bf16_t* __restrict__ m2bf, const bf16_t* __restrict__ xbf,
                       const bf16_t* __restrict__ x2bf, const bf16_t* __restrict__ xT,
                       const float* __restrict__ w, const float* __restrict__ c,
                       const float* __restrict__ sumqp, const float* __restrict__ nb2,
                       bf16_t* __restrict__ qrbf) {
  __shared__ __align__(16) bf16_t sc[4][16 * 32];
  const int lane = threadIdx.x & 31;
  const int wv   = threadIdx.x >> 5;
  const int half = lane >> 4;
  const int lcol = lane & 15;
  const int tile = blockIdx.x * 4 + wv;  // 0..511
  const int b    = tile >> 6;            // 64 tiles per batch
  const int i0   = (tile & 63) * 16;
  const long pos0 = (long)b * S_ + i0;
  bf16_t* smem = sc[wv];

  float wi[8], ci[8], sq[8], nbv[8];
#pragma unroll
  for (int r = 0; r < 8; ++r) {
    const long row = pos0 + r + 8 * half;
    wi[r] = w[row]; ci[r] = c[row]; sq[r] = sumqp[row]; nbv[r] = sqrtf(nb2[row]);
  }

  const bf16_t* r1 = a1  + (size_t)(pos0 + lcol) * D_;
  const bf16_t* r2 = mbf + (size_t)(pos0 + lcol) * D_;
  const bf16_t* r3 = m2bf + (size_t)(pos0 + lcol) * D_;
  v16bf Aa[4], Am[4], A2[4];
#pragma unroll
  for (int kc = 0; kc < 4; ++kc) {
    Aa[kc] = load_a_frag(r1, kc * 32, half);
    Am[kc] = load_a_frag(r2, kc * 32, half);
    A2[kc] = load_a_frag(r3, kc * 32, half);
  }

  v8f Cqr[8];
#pragma unroll
  for (int nt = 0; nt < 8; ++nt) Cqr[nt] = zero8();

  for (int j0 = 0; j0 < i0 + 16; j0 += 32) {
    v8f T1[2], S1[2], S2v[2];
#pragma unroll
    for (int st = 0; st < 2; ++st) { T1[st] = zero8(); S1[st] = zero8(); S2v[st] = zero8(); }

#pragma unroll
    for (int st = 0; st < 2; ++st) {
      const size_t jpos = (size_t)b * S_ + j0 + st * 16 + lcol;
      const bf16_t* xr  = xbf  + jpos * D_;
      const bf16_t* x2r = x2bf + jpos * D_;
#pragma unroll
      for (int kc = 0; kc < 4; ++kc) {
        const int k0 = kc * 32 + 16 * half;
        v16bf Bx = load_b_frag(xr + k0);
        T1[st]  = wmma_bf16(Aa[kc], Bx, T1[st]);
        S1[st]  = wmma_bf16(Am[kc], Bx, S1[st]);
        S2v[st] = wmma_bf16(A2[kc], load_b_frag(x2r + k0), S2v[st]);
      }
    }

    // pe/angle math on C-layout tiles, park bf16 scores in LDS
#pragma unroll
    for (int st = 0; st < 2; ++st) {
      const int jg = j0 + st * 16 + lcol;
#pragma unroll
      for (int r = 0; r < 8; ++r) {
        const int il = r + 8 * half;
        const float dist = (float)(i0 + il - jg);
        float score = 0.f;
        if (dist > 0.f) {
          const float pe  = 1.f / coshf((dist - ci[r]) / (wi[r] + EPS_));
          const float dot = pe * T1[st][r] + EPS_ * sq[r];
          const float na  = sqrtf(pe * pe * S2v[st][r] + 2.f * EPS_ * pe * S1[st][r] +
                                  (float)D_ * EPS_ * EPS_);
          float cs = dot / (na * nbv[r]);
          cs = fminf(1.f, fmaxf(-1.f, cs));
          score = pe * acosf(cs);
        }
        smem[il * 32 + st * 16 + lcol] = (bf16_t)score;
      }
    }
    __builtin_amdgcn_wave_barrier();
    asm volatile("s_wait_dscnt 0" ::: "memory");  // wave-local LDS C->A relayout

    v16bf As = load_a_frag(smem + lcol * 32, 0, half);
    __builtin_amdgcn_wave_barrier();

    // qr[i,d] += score[i,j] * x[j,d]  (B from xT so K=j is contiguous)
#pragma unroll
    for (int nt = 0; nt < 8; ++nt) {
      const bf16_t* bcol = xT + ((size_t)b * D_ + nt * 16 + lcol) * S_ + j0 + 16 * half;
      Cqr[nt] = wmma_bf16(As, load_b_frag(bcol), Cqr[nt]);
    }
  }

#pragma unroll
  for (int nt = 0; nt < 8; ++nt)
#pragma unroll
    for (int r = 0; r < 8; ++r)
      qrbf[(size_t)(pos0 + r + 8 * half) * D_ + nt * 16 + lcol] = (bf16_t)Cqr[nt][r];
}

// ---------------- emb_map contraction + residual + LayerNorm ----------------
// mapped[p,d] = sum_{j,k} x[p,j] qr[p,k] map[d,j,k].
// TDM double-buffers 8 KB map tiles (K=32 x N=128) into LDS, shared by 4 waves;
// A is built on the fly as x_j * qr-slice with packed bf16 multiplies.
__global__ void k_emb(const bf16_t* __restrict__ xbf, const bf16_t* __restrict__ qrbf,
                      const bf16_t* __restrict__ mapbf, const float* __restrict__ x,
                      const float* __restrict__ gamma, const float* __restrict__ beta,
                      float* __restrict__ out) {
  __shared__ __align__(64) bf16_t stage[2][D_ * 32];  // 2 x 8 KB, LDS offsets 0 / 8192
  const int lane = threadIdx.x & 31;
  const int wv   = threadIdx.x >> 5;
  const int half = lane >> 4;
  const int lcol = lane & 15;
  const long pos0 = (long)(blockIdx.x * 4 + wv) * 16;

  const bf16_t* xrow = xbf  + (size_t)(pos0 + lcol) * D_;
  const bf16_t* qrow = qrbf + (size_t)(pos0 + lcol) * D_;

  v16bf Aq[4];
#pragma unroll
  for (int kc = 0; kc < 4; ++kc) Aq[kc] = load_a_frag(qrow, kc * 32, half);

  v8f C[8];
#pragma unroll
  for (int nt = 0; nt < 8; ++nt) C[nt] = zero8();

#if HAVE_TDM
  if (wv == 0) tdm_load_tile(mapbf, 0u);
#endif

  for (int ch = 0; ch < 512; ++ch) {   // chunk ch covers k_lin = [ch*32, ch*32+32)
    const int j  = ch >> 2;
    const int kc = ch & 3;
#if HAVE_TDM
    if (wv == 0) {
      if (ch + 1 < 512) {
        // prefetch next tile into the other buffer (read last at iter ch-1,
        // all readers released by the trailing barrier of iter ch-1)
        tdm_load_tile(mapbf + (size_t)(ch + 1) * 32,
                      ((ch + 1) & 1) ? (unsigned)(D_ * 32 * 2) : 0u);
        __builtin_amdgcn_s_wait_tensorcnt((short)1);  // tile ch has landed
      } else {
        __builtin_amdgcn_s_wait_tensorcnt((short)0);
      }
    }
    __syncthreads();  // tile ch visible to all 4 waves
#else
    __syncthreads();  // previous tile fully consumed
    {
      const uint32_t* gs =
          (const uint32_t*)(mapbf + (size_t)ch * 32 + (size_t)threadIdx.x * (D_ * D_));
      uint32_t* ds = (uint32_t*)(stage[ch & 1] + threadIdx.x * 32);
#pragma unroll
      for (int u = 0; u < 16; ++u) ds[u] = gs[u];
    }
    __syncthreads();
#endif
    const bf16_t* sb = stage[ch & 1];
    const v16bf A = mul16(Aq[kc], xrow[j]);  // v_pk_mul_bf16 scale by x[row][j]
#pragma unroll
    for (int nt = 0; nt < 8; ++nt) {
      C[nt] = wmma_bf16(A, load_b_frag(sb + (nt * 16 + lcol) * 32 + 16 * half), C[nt]);
    }
#if HAVE_TDM
    __syncthreads();  // all readers done before this buffer is re-filled
#endif
  }

  // residual
#pragma unroll
  for (int nt = 0; nt < 8; ++nt) {
    const int col = nt * 16 + lcol;
#pragma unroll
    for (int r = 0; r < 8; ++r)
      C[nt][r] += x[(size_t)(pos0 + r + 8 * half) * D_ + col];
  }
  // LayerNorm stats via 16-lane butterfly per half
  float mu[8], rs[8];
#pragma unroll
  for (int r = 0; r < 8; ++r) {
    float s = 0.f, s2 = 0.f;
#pragma unroll
    for (int nt = 0; nt < 8; ++nt) { const float v = C[nt][r]; s += v; s2 += v * v; }
#pragma unroll
    for (int msk = 1; msk <= 8; msk <<= 1) {
      s  += __shfl_xor(s, msk, 32);
      s2 += __shfl_xor(s2, msk, 32);
    }
    mu[r] = s * (1.f / (float)D_);
    const float var = s2 * (1.f / (float)D_) - mu[r] * mu[r];
    rs[r] = rsqrtf(var + LN_EPS_);
  }
#pragma unroll
  for (int nt = 0; nt < 8; ++nt) {
    const int col = nt * 16 + lcol;
    const float g = gamma[col], be = beta[col];
#pragma unroll
    for (int r = 0; r < 8; ++r)
      out[(size_t)(pos0 + r + 8 * half) * D_ + col] =
          (C[nt][r] - mu[r]) * rs[r] * g + be;
  }
}

// ---------------- launch ----------------

extern "C" void kernel_launch(void* const* d_in, const int* in_sizes, int n_in,
                              void* d_out, int out_size, void* d_ws, size_t ws_size,
                              hipStream_t stream) {
  (void)in_sizes; (void)n_in; (void)out_size; (void)ws_size;
  const float* x    = (const float*)d_in[0];
  const float* Wq   = (const float*)d_in[1];
  const float* bq   = (const float*)d_in[2];
  const float* Wm   = (const float*)d_in[3];
  const float* bm   = (const float*)d_in[4];
  const float* Ww   = (const float*)d_in[5];
  const float* bw   = (const float*)d_in[6];
  const float* Wc   = (const float*)d_in[7];
  const float* bc   = (const float*)d_in[8];
  const float* emb  = (const float*)d_in[9];
  const float* gamm = (const float*)d_in[10];
  const float* beta = (const float*)d_in[11];
  float* out = (float*)d_out;

  char* ws = (char*)d_ws;
  size_t off = 0;
  auto take = [&](size_t bytes) -> char* {
    char* p = ws + off;
    off = (off + bytes + 255) & ~(size_t)255;
    return p;
  };
  bf16_t* x_bf   = (bf16_t*)take((size_t)P_ * D_ * 2);
  bf16_t* x2_bf  = (bf16_t*)take((size_t)P_ * D_ * 2);
  bf16_t* xT_bf  = (bf16_t*)take((size_t)P_ * D_ * 2);
  bf16_t* a1_bf  = (bf16_t*)take((size_t)P_ * D_ * 2);
  bf16_t* m_bf   = (bf16_t*)take((size_t)P_ * D_ * 2);
  bf16_t* m2_bf  = (bf16_t*)take((size_t)P_ * D_ * 2);
  bf16_t* qr_bf  = (bf16_t*)take((size_t)P_ * D_ * 2);
  bf16_t* map_bf = (bf16_t*)take((size_t)D_ * D_ * D_ * 2);
  bf16_t* WqT    = (bf16_t*)take((size_t)D_ * D_ * 2);
  bf16_t* WmT    = (bf16_t*)take((size_t)D_ * D_ * 2);
  float*  wv     = (float*)take((size_t)P_ * 4);
  float*  cv     = (float*)take((size_t)P_ * 4);
  float*  sumqp  = (float*)take((size_t)P_ * 4);
  float*  nb2    = (float*)take((size_t)P_ * 4);

  hipMemsetAsync(sumqp, 0, (size_t)P_ * 4, stream);
  hipMemsetAsync(nb2,   0, (size_t)P_ * 4, stream);

  k_cast_x<<<(P_ * D_) / 256, 256, 0, stream>>>(x, x_bf, x2_bf, xT_bf);
  k_cast_w<<<(D_ * D_) / 256, 256, 0, stream>>>(Wq, Wm, WqT, WmT);
  k_cast_map<<<(D_ * D_ * D_) / 256, 256, 0, stream>>>(emb, map_bf);
  k_wc<<<P_ / 256, 256, 0, stream>>>(x, Ww, bw, Wc, bc, wv, cv);
  k_qm<<<P_ / 16, 128, 0, stream>>>(bq, bm, x_bf, WqT, WmT, a1_bf, m_bf, m2_bf,
                                    sumqp, nb2);
  k_attn<<<P_ / 64, 128, 0, stream>>>(a1_bf, m_bf, m2_bf, x_bf, x2_bf, xT_bf, wv, cv,
                                      sumqp, nb2, qr_bf);
  k_emb<<<P_ / 64, 128, 0, stream>>>(x_bf, qr_bf, map_bf, x, gamm, beta, out);
}
